// MultiHeadSelfAttention_46986942218895
// MI455X (gfx1250) — matrix operand order repaired and checked by reading, not compiled, not run on hardware
//
#include <hip/hip_runtime.h>

// ---------------------------------------------------------------------------
// MHA for MI455X (gfx1250): bf16 WMMA + fp32 accumulate.
// GEMM tiles staged by the Tensor Data Mover (TENSORcnt, hw LDS padding),
// attention K tile staged by async global->LDS (ASYNCcnt), softmax
// reductions in VALU via DPP. Double-buffered LDS everywhere.
// B=2, S=2048, D=1024, H=16, HD=64.
// ---------------------------------------------------------------------------

typedef __bf16 bf16_t;
typedef __bf16 bf16x8 __attribute__((ext_vector_type(8)));
typedef __bf16 v16bf  __attribute__((ext_vector_type(16)));
typedef float  v8f    __attribute__((ext_vector_type(8)));
typedef unsigned int v4u __attribute__((ext_vector_type(4)));
typedef int v4i __attribute__((ext_vector_type(4)));
typedef int v8i __attribute__((ext_vector_type(8)));

__device__ __forceinline__ v8f wmma_bf16(v16bf a, v16bf b, v8f c) {
  return __builtin_amdgcn_wmma_f32_16x16x32_bf16(
      /*neg_a=*/false, a, /*neg_b=*/false, b,
      /*c_mod=*/(short)0, c, /*reuse_a=*/false, /*reuse_b=*/false);
}

// Build a v16bf fragment from two 16-byte (8 x bf16) chunks.
__device__ __forceinline__ v16bf make_frag(const bf16_t* lo, const bf16_t* hi) {
  bf16x8 a = *(const bf16x8*)lo;
  bf16x8 b = *(const bf16x8*)hi;
  v16bf r;
#pragma unroll
  for (int i = 0; i < 8; ++i) { r[i] = a[i]; r[8 + i] = b[i]; }
  return r;
}

// ---- async global->LDS (ASYNCcnt path) -----------------------------------
__device__ __forceinline__ void async_b128(const bf16_t* lds_dst,
                                           const bf16_t* gsrc) {
  uint32_t l = (uint32_t)(uintptr_t)lds_dst;   // low 32 bits = LDS offset
  uint64_t g = (uint64_t)(uintptr_t)gsrc;
  asm volatile("global_load_async_to_lds_b128 %0, %1, off"
               :: "v"(l), "v"(g) : "memory");
}
__device__ __forceinline__ void async_b128x2(const bf16_t* lds_dst,
                                             const bf16_t* gsrc) {
  uint32_t l = (uint32_t)(uintptr_t)lds_dst;
  uint64_t g = (uint64_t)(uintptr_t)gsrc;
  // inst offset applies to BOTH the LDS and global address
  asm volatile("global_load_async_to_lds_b128 %0, %1, off\n\t"
               "global_load_async_to_lds_b128 %0, %1, off offset:16"
               :: "v"(l), "v"(g) : "memory");
}
__device__ __forceinline__ void wait_asynccnt0() {
#if __has_builtin(__builtin_amdgcn_s_wait_asynccnt)
  __builtin_amdgcn_s_wait_asynccnt(0);
#else
  asm volatile("s_wait_asynccnt 0x0" ::: "memory");
#endif
}

// ---- Tensor Data Mover: 2D bf16 tile, global -> LDS (TENSORcnt path) ------
// Descriptor per CDNA5 ISA 8.3/8.4. tensor_dim == tile_dim (stride may exceed
// dims), pad_enable inserts LDS padding after every tile row in hardware.
__device__ __forceinline__ void tdm_load_2d_bf16(
    uint32_t lds_addr, const bf16_t* gaddr, uint32_t tile0, uint32_t tile1,
    uint32_t stride0_elems, uint32_t pad_interval_code,
    uint32_t pad_amount_code) {
#if __has_builtin(__builtin_amdgcn_tensor_load_to_lds)
  const uint64_t ga = (uint64_t)(uintptr_t)gaddr;
  v4u g0;
  g0.x = 1u;                                  // count=1, user mode, no gather
  g0.y = lds_addr;                            // bits 63:32  lds_addr
  g0.z = (uint32_t)ga;                        // bits 95:64  global_addr lo
  g0.w = (uint32_t)(ga >> 32) | (2u << 30);   // addr[56:32] | type=2 (image)
  v8i g1;
  g1[0] = (int)((1u << 16)                    // data_size = 2 bytes
                | (1u << 20)                  // pad_enable
                | (pad_interval_code << 22) | (pad_amount_code << 25));
  g1[1] = (int)((tile0 & 0xFFFFu) << 16);     // tensor_dim0[15:0]
  g1[2] = (int)((tile0 >> 16) | ((tile1 & 0xFFFFu) << 16));  // d0 hi | d1 lo
  g1[3] = (int)((tile1 >> 16) | (tile0 << 16));              // d1 hi | tile_dim0
  g1[4] = (int)(tile1 & 0xFFFFu);             // tile_dim1 (tile_dim2 = 0)
  g1[5] = (int)stride0_elems;                 // tensor_dim0_stride lo
  g1[6] = 0;
  g1[7] = 0;
  v4i z4 = {0, 0, 0, 0};
#if defined(__clang_major__) && (__clang_major__ >= 23)
  v8i z8 = {0, 0, 0, 0, 0, 0, 0, 0};
  __builtin_amdgcn_tensor_load_to_lds(g0, g1, z4, z4, z8, 0);
#else
  __builtin_amdgcn_tensor_load_to_lds(g0, g1, z4, z4, 0);
#endif
#else
  // Fallback: should not happen on probed toolchains.
  (void)lds_addr; (void)gaddr; (void)tile0; (void)tile1;
  (void)stride0_elems; (void)pad_interval_code; (void)pad_amount_code;
#endif
}
__device__ __forceinline__ void wait_tensorcnt0() {
#if __has_builtin(__builtin_amdgcn_s_wait_tensorcnt)
  __builtin_amdgcn_s_wait_tensorcnt(0);
#else
  asm volatile("s_wait_tensorcnt 0x0" ::: "memory");
#endif
}

// ---- DPP 16-lane butterfly reductions (no LDS traffic) --------------------
template <int CTRL>
__device__ __forceinline__ float dpp_movf(float x) {
  return __builtin_bit_cast(
      float, __builtin_amdgcn_update_dpp(0, __builtin_bit_cast(int, x), CTRL,
                                         0xf, 0xf, true));
}
__device__ __forceinline__ float rmax16(float x) {
  x = fmaxf(x, dpp_movf<0xB1>(x));   // quad_perm(1,0,3,2): xor1
  x = fmaxf(x, dpp_movf<0x4E>(x));   // quad_perm(2,3,0,1): xor2
  x = fmaxf(x, dpp_movf<0x141>(x));  // row_half_mirror: folds 8-groups
  x = fmaxf(x, dpp_movf<0x140>(x));  // row_mirror: folds 16-groups
  return x;
}
__device__ __forceinline__ float rsum16(float x) {
  x += dpp_movf<0xB1>(x);
  x += dpp_movf<0x4E>(x);
  x += dpp_movf<0x141>(x);
  x += dpp_movf<0x140>(x);
  return x;
}

// ---------------------------------------------------------------------------
// Converters (one-shot, bandwidth-trivial)
// ---------------------------------------------------------------------------
__global__ void cvt_f32_bf16(const float* __restrict__ in,
                             bf16_t* __restrict__ out, int n) {
  int i = blockIdx.x * blockDim.x + threadIdx.x;
  if (i < n) out[i] = (bf16_t)in[i];
}

// in [R][C] fp32 (row-major) -> out [C][R] bf16 (row-major), i.e. transpose.
__global__ void cvt_transpose_bf16(const float* __restrict__ in,
                                   bf16_t* __restrict__ out, int R, int C) {
  int i = blockIdx.x * blockDim.x + threadIdx.x;
  if (i < R * C) {
    int c = i / R;
    int r = i - c * R;
    out[(size_t)c * R + r] = (bf16_t)in[(size_t)r * C + c];
  }
}

// ---------------------------------------------------------------------------
// GEMM: C[M][N] = A[M][K] (bf16) * Bt[N][K]^T (bf16) + bias[N]
// 256 threads = 8 waves; block tile 128x128, BK=32; wave tile 64x32.
// TDM-staged, double-buffered LDS: wave0 issues 2 tensor_load_to_lds per
// K-step (hw-padded to PITCH=40), one barrier per K-step.
// ---------------------------------------------------------------------------
template <bool OUT_BF16>
__global__ __launch_bounds__(256) void gemm_bf16_kernel(
    const bf16_t* __restrict__ A, const bf16_t* __restrict__ Bt,
    const float* __restrict__ bias, float* __restrict__ Cf,
    bf16_t* __restrict__ Cb, int M, int N, int K) {
  constexpr int BM = 128, BN = 128, BK = 32, PITCH = 40;  // 80B rows (64+16)
  __shared__ __align__(16) bf16_t sA[2][BM * PITCH];
  __shared__ __align__(16) bf16_t sB[2][BN * PITCH];

  const int tid  = threadIdx.x;
  const int lane = tid & 31;
  const int wave = tid >> 5;
  const int wm   = wave & 1;   // 0..1  -> 64-row slab
  const int wn   = wave >> 1;  // 0..3  -> 32-col slab
  const int l16  = lane & 15;
  const int lh   = lane >> 4;  // half-wave select

  const int blockM = blockIdx.y * BM;
  const int blockN = blockIdx.x * BN;

  v8f z = {};
  v8f acc[4][2];
#pragma unroll
  for (int mi = 0; mi < 4; ++mi)
#pragma unroll
    for (int ni = 0; ni < 2; ++ni) acc[mi][ni] = z;

  // TDM stage: one 128x32 bf16 tile for A and one for B per K-step.
  // pad_interval=3 (16 DW = 64B = one tile row), pad_amount=3 (4 DW = 16B)
  // -> LDS pitch 80B = PITCH bf16.
  auto stage = [&](int buf, int k0) {
    if (wave == 0) {
      tdm_load_2d_bf16((uint32_t)(uintptr_t)&sA[buf][0],
                       A + (size_t)blockM * K + k0, BK, BM, (uint32_t)K, 3, 3);
      tdm_load_2d_bf16((uint32_t)(uintptr_t)&sB[buf][0],
                       Bt + (size_t)blockN * K + k0, BK, BN, (uint32_t)K, 3, 3);
    }
  };

  const int NK = K / BK;
  stage(0, 0);
  wait_tensorcnt0();
  __syncthreads();

  for (int kb = 0; kb < NK; ++kb) {
    const int cur = kb & 1;
    if (kb + 1 < NK) stage(cur ^ 1, (kb + 1) * BK);

    v16bf af[4], bfr[2];
#pragma unroll
    for (int mi = 0; mi < 4; ++mi) {
      const bf16_t* p = &sA[cur][(wm * 64 + mi * 16 + l16) * PITCH + lh * 8];
      af[mi] = make_frag(p, p + 16);   // K = lh*8+e%8 (+16 for hi chunk)
    }
#pragma unroll
    for (int ni = 0; ni < 2; ++ni) {
      const bf16_t* p = &sB[cur][(wn * 32 + ni * 16 + l16) * PITCH + lh * 8];
      bfr[ni] = make_frag(p, p + 16);  // K = lh*8+e%8 (+16 for hi chunk)
    }
#pragma unroll
    for (int mi = 0; mi < 4; ++mi)
#pragma unroll
      for (int ni = 0; ni < 2; ++ni)
        acc[mi][ni] = wmma_bf16(af[mi], bfr[ni], acc[mi][ni]);

    wait_tensorcnt0();  // wave0 drains its TDM ops; others are trivially 0
    __syncthreads();    // everyone done reading cur / tile nxt landed
  }

  // Epilogue: C layout row M = lh*8+v, col N = l16.
#pragma unroll
  for (int mi = 0; mi < 4; ++mi)
#pragma unroll
    for (int ni = 0; ni < 2; ++ni) {
      const int col = blockN + wn * 32 + ni * 16 + l16;
      const float bv = bias[col];
#pragma unroll
      for (int v = 0; v < 8; ++v) {
        const int row = blockM + wm * 64 + mi * 16 + lh * 8 + v;
        const float val = acc[mi][ni][v] + bv;
        if constexpr (OUT_BF16) Cb[(size_t)row * N + col] = (bf16_t)val;
        else                    Cf[(size_t)row * N + col] = val;
      }
    }
}

// ---------------------------------------------------------------------------
// Flash attention: one workgroup = 128 q-rows of one (b,h); 8 waves x 16 rows.
// qkv: [B*S][3*D] bf16 (Q at +0, K at +D, V at +2D within a row).
// out: [B*S][D]   bf16 (heads re-interleaved).
// K tile staged with async global->LDS copies; K/V double-buffered.
// ---------------------------------------------------------------------------
__global__ __launch_bounds__(256) void attn_kernel(
    const bf16_t* __restrict__ qkv, bf16_t* __restrict__ out) {
  constexpr int S = 2048, D = 1024, HD = 64;
  constexpr int KP = 72;  // sK pitch (bf16), 144B rows
  constexpr int VP = 40;  // sVt pitch
  constexpr int PP = 40;  // sP pitch
  __shared__ __align__(16) bf16_t sK[2][32 * KP];      // [key][feature]
  __shared__ __align__(16) bf16_t sVt[2][64 * VP];     // [feature][key]
  __shared__ __align__(16) bf16_t sP[8 * 16 * PP];     // per-wave [row][key]

  const int tid  = threadIdx.x;
  const int lane = tid & 31;
  const int wave = tid >> 5;
  const int l16  = lane & 15;
  const int lh   = lane >> 4;

  const int bh = blockIdx.y;
  const int b  = bh >> 4;
  const int h  = bh & 15;
  const int qbase = blockIdx.x * 128 + wave * 16;

  // Q fragments (held in VGPRs for the whole kernel): rows qbase..qbase+15.
  const size_t rowQ = (size_t)(b * S + qbase + l16) * (3 * D) + h * HD;
  v16bf qf[2];
#pragma unroll
  for (int fb = 0; fb < 2; ++fb) {
    const bf16_t* p = qkv + rowQ + fb * 32 + lh * 8;
    qf[fb] = make_frag(p, p + 16);
  }

  float m_[8], l_[8];
#pragma unroll
  for (int v = 0; v < 8; ++v) { m_[v] = -INFINITY; l_[v] = 0.0f; }
  v8f z = {};
  v8f o[4] = {z, z, z, z};

  const float scale = 0.125f;  // 1/sqrt(64)

  // Cooperative loaders: 256 threads x 8 bf16 = one 32x64 tile.
  const int ldKey = tid >> 3;        // 0..31
  const int ldF   = (tid & 7) * 8;   // 0,8,...,56
  bf16_t* myP = &sP[wave * 16 * PP];

  auto load_kv = [&](int buf, int kb) {
    const size_t srcRow = (size_t)(b * S + kb * 32 + ldKey) * (3 * D) + h * HD;
    // K: async global->LDS, row-major (features contiguous)
    async_b128(&sK[buf][ldKey * KP + ldF], qkv + srcRow + D + ldF);
    // V: through VGPRs, transposed into LDS (keys contiguous per feature)
    bf16x8 vvec = *(const bf16x8*)(qkv + srcRow + 2 * D + ldF);
#pragma unroll
    for (int i = 0; i < 8; ++i) sVt[buf][(ldF + i) * VP + ldKey] = vvec[i];
  };

  const int NB = S / 32;
  load_kv(0, 0);
  wait_asynccnt0();
  __syncthreads();

  for (int kb = 0; kb < NB; ++kb) {
    const int cur = kb & 1;
    if (kb + 1 < NB) load_kv(cur ^ 1, kb + 1);

    // Load all 4 K fragments up front so ds_loads batch/overlap with WMMA.
    v16bf kf[4];
#pragma unroll
    for (int fb = 0; fb < 2; ++fb) {
      const bf16_t* p0 = &sK[cur][l16 * KP + fb * 32 + lh * 8];
      kf[fb] = make_frag(p0, p0 + 16);         // keys 0..15
      const bf16_t* p1 = &sK[cur][(16 + l16) * KP + fb * 32 + lh * 8];
      kf[2 + fb] = make_frag(p1, p1 + 16);     // keys 16..31
    }
    // scores: s0 = keys kb*32..+15, s1 = keys +16..+31 (K-dim = 64 features)
    v8f s0 = z, s1 = z;
    s0 = wmma_bf16(qf[0], kf[0], s0);
    s1 = wmma_bf16(qf[0], kf[2], s1);
    s0 = wmma_bf16(qf[1], kf[1], s0);
    s1 = wmma_bf16(qf[1], kf[3], s1);

    // online softmax (row = lh*8+v, columns striped across 16 lanes); all
    // cross-lane reductions in VALU via DPP.
    float alph[8];
#pragma unroll
    for (int v = 0; v < 8; ++v) {
      const float a0 = s0[v] * scale;
      const float a1 = s1[v] * scale;
      const float bm = rmax16(fmaxf(a0, a1));
      const float mn = fmaxf(m_[v], bm);
      const float al = __expf(m_[v] - mn);
      const float p0 = __expf(a0 - mn);
      const float p1 = __expf(a1 - mn);
      l_[v] = l_[v] * al + rsum16(p0 + p1);
      m_[v] = mn;
      alph[v] = al;
      myP[(lh * 8 + v) * PP + l16]      = (bf16_t)p0;
      myP[(lh * 8 + v) * PP + 16 + l16] = (bf16_t)p1;
    }
    __builtin_amdgcn_wave_barrier();  // keep C->A restripe store/load ordered

    // rescale running output
#pragma unroll
    for (int t = 0; t < 4; ++t)
#pragma unroll
      for (int v = 0; v < 8; ++v) o[t][v] *= alph[v];

    // P fragment (A layout): row = l16, keys = lh*8+e%8 (+16 hi chunk)
    const bf16_t* pp = &myP[l16 * PP + lh * 8];
    const v16bf pf = make_frag(pp, pp + 16);

    // o += P(16x32) * V(32x64): B frag col = feature, keys = lh*8+e%8 (+16 hi chunk)
    v16bf vf[4];
#pragma unroll
    for (int t = 0; t < 4; ++t) {
      const bf16_t* vp = &sVt[cur][(t * 16 + l16) * VP + lh * 8];
      vf[t] = make_frag(vp, vp + 16);
    }
#pragma unroll
    for (int t = 0; t < 4; ++t) o[t] = wmma_bf16(pf, vf[t], o[t]);

    wait_asynccnt0();
    __syncthreads();
  }

  // normalize + write attn output (bf16) with heads re-interleaved
#pragma unroll
  for (int v = 0; v < 8; ++v) {
    const float inv = 1.0f / l_[v];
    const int row = b * S + qbase + lh * 8 + v;
#pragma unroll
    for (int t = 0; t < 4; ++t) {
      out[(size_t)row * D + h * HD + t * 16 + l16] = (bf16_t)(o[t][v] * inv);
    }
  }
}

// ---------------------------------------------------------------------------
// Launcher
// ---------------------------------------------------------------------------
extern "C" void kernel_launch(void* const* d_in, const int* in_sizes, int n_in,
                              void* d_out, int out_size, void* d_ws,
                              size_t ws_size, hipStream_t stream) {
  (void)in_sizes; (void)n_in; (void)out_size; (void)ws_size;
  constexpr int B = 2, S = 2048, D = 1024;
  const int M = B * S;  // 4096

  const float* x     = (const float*)d_in[0];
  const float* w_qkv = (const float*)d_in[1];
  const float* b_qkv = (const float*)d_in[2];
  const float* w_out = (const float*)d_in[3];
  const float* b_out = (const float*)d_in[4];
  float* outp        = (float*)d_out;

  char* ws = (char*)d_ws;
  bf16_t* xb    = (bf16_t*)ws; ws += (size_t)M * D * 2;          //  8 MB
  bf16_t* wqkvT = (bf16_t*)ws; ws += (size_t)3 * D * D * 2;      //  6 MB
  bf16_t* woutT = (bf16_t*)ws; ws += (size_t)D * D * 2;          //  2 MB
  bf16_t* qkv   = (bf16_t*)ws; ws += (size_t)M * 3 * D * 2;      // 25 MB
  bf16_t* attn  = (bf16_t*)ws;                                   //  8 MB

  {
    int n = M * D;
    cvt_f32_bf16<<<(n + 255) / 256, 256, 0, stream>>>(x, xb, n);
  }
  {
    int n = 3 * D * D;  // w_qkv [D][3D] -> wqkvT [3D][D]
    cvt_transpose_bf16<<<(n + 255) / 256, 256, 0, stream>>>(w_qkv, wqkvT, D, 3 * D);
  }
  {
    int n = D * D;      // w_out [D][D] -> woutT [D][D]
    cvt_transpose_bf16<<<(n + 255) / 256, 256, 0, stream>>>(w_out, woutT, D, D);
  }

  // qkv = x @ w_qkv + b_qkv  (bf16 out)
  gemm_bf16_kernel<true><<<dim3((3 * D) / 128, M / 128), 256, 0, stream>>>(
      xb, wqkvT, b_qkv, nullptr, qkv, M, 3 * D, D);

  // attention per (b,h), 128 q-rows per workgroup
  attn_kernel<<<dim3(S / 128, B * 16), 256, 0, stream>>>(qkv, attn);

  // out = attn @ w_out + b_out  (fp32 out)
  gemm_bf16_kernel<false><<<dim3(D / 128, M / 128), 256, 0, stream>>>(
      attn, woutT, b_out, outp, nullptr, M, D, D);
}
